// CNN_47004122087523
// MI455X (gfx1250) — matrix operand order repaired and checked by reading, compiled, not run on hardware
//
#include <hip/hip_runtime.h>
#include <hip/hip_bf16.h>

typedef __bf16 bf16;
typedef __attribute__((ext_vector_type(16))) __bf16 v16bf;
typedef __attribute__((ext_vector_type(8)))  float  v8f;
typedef __attribute__((ext_vector_type(4)))  int    v4i;

#if defined(__has_builtin)
#  if __has_builtin(__builtin_amdgcn_global_load_async_to_lds_b128)
#    define HAS_ASYNC_LDS 1
#  endif
#endif
#ifndef HAS_ASYNC_LDS
#  define HAS_ASYNC_LDS 0
#  warning "__builtin_amdgcn_global_load_async_to_lds_b128 not available; using direct global loads for A fragments"
#endif

static __device__ __forceinline__ void wait_async0()
{
#if defined(__has_builtin) && __has_builtin(__builtin_amdgcn_s_wait_asynccnt)
    __builtin_amdgcn_s_wait_asynccnt(0);
#else
    asm volatile("s_wait_asynccnt 0x0" ::: "memory");
#endif
}

#if HAS_ASYNC_LDS
static __device__ __forceinline__ void async_copy16(const uint4* gsrc, uint4* ldst)
{
    // builtin takes generic int4* params; clang inserts the addrspace casts
    __builtin_amdgcn_global_load_async_to_lds_b128((v4i*)gsrc, (v4i*)ldst, 0, 0);
}
#endif

// ---------------------------------------------------------------------------
// Weight prepack: f32 (Cout,Cin,KH,KW) -> bf16 WMMA A-fragment layout.
// A-matrix 16x32 (MxK), 16-bit elements (ISA 7.12.2):
//   lane L: m = L&15, kbase = (L>>4)*8
//   vgpr i<4 : k = kbase + 2i, 2i+1 ; vgpr i>=4 : k = kbase + 16 + 2(i-4), +1
// Storage: Apack[((mt*KT + kt)*32 + lane)*8 + i]  (u32 = packed bf16 pair)
// ---------------------------------------------------------------------------
__global__ void pack_weights(const float* __restrict__ W, unsigned* __restrict__ Apack,
                             int Cout, int K, int MT, int KT)
{
    int idx = blockIdx.x * blockDim.x + threadIdx.x;
    int total = MT * KT * 32;
    if (idx >= total) return;
    int lane = idx & 31;
    int t    = idx >> 5;
    int kt   = t % KT;
    int mt   = t / KT;
    int m     = mt * 16 + (lane & 15);
    int kbase = kt * 32 + ((lane >> 4) * 8);

    unsigned out[8];
    for (int i = 0; i < 8; ++i) {
        int k0 = kbase + ((i < 4) ? 2 * i : 16 + 2 * (i - 4));
        unsigned short h[2];
        for (int u = 0; u < 2; ++u) {
            int k = k0 + u;
            float v = 0.0f;
            if (m < Cout && k < K) v = W[(size_t)m * K + k];
            union { bf16 b; unsigned short s; } cv;
            cv.b = (bf16)v;
            h[u] = cv.s;
        }
        out[i] = (unsigned)h[0] | ((unsigned)h[1] << 16);
    }
    unsigned* dst = Apack + (size_t)idx * 8;
    for (int i = 0; i < 8; ++i) dst[i] = out[i];
}

// ---------------------------------------------------------------------------
// Build c1 input: windowed view of x (B=8,T=128,F=144) -> (1024,1,144,15) bf16
// ---------------------------------------------------------------------------
__global__ void build_window(const float* __restrict__ x, bf16* __restrict__ out, int total)
{
    int idx = blockIdx.x * blockDim.x + threadIdx.x;
    if (idx >= total) return;
    int w  = idx % 15;
    int t2 = idx / 15;
    int f  = t2 % 144;
    int n  = t2 / 144;
    int b  = n / 128, t = n % 128;
    int tt = t + w - 7;
    float v = (tt >= 0 && tt < 128) ? x[((size_t)b * 128 + tt) * 144 + f] : 0.0f;
    out[idx] = (bf16)v;
}

// ---------------------------------------------------------------------------
// Implicit-GEMM conv + fused BN/ReLU epilogue, bf16 WMMA (16x16x32, f32 acc).
// All geometry compile-time. Block: 256 threads = 8 waves, tile MTB*16 x 64.
// Per K-step: waves 0..MTB-1 async-DMA their 1KB A m-tile into LDS
// (global_load_async_to_lds_b128, ASYNCcnt) overlapping the im2col gather of
// the 32x64 B chunk; after barrier all waves read A/B fragments from LDS.
// FLAGS: bit0 = BN, bit1 = ReLU, bit2 = store f32 (else bf16)
// ---------------------------------------------------------------------------
template <int Cin, int H, int W, int Cout, int KH, int KW, int PAD, int MTB, int FLAGS>
__global__ __launch_bounds__(256) void conv_wmma(
    const bf16* __restrict__ in, const unsigned* __restrict__ Apack,
    bf16* __restrict__ outb, float* __restrict__ outf,
    const float* __restrict__ bias, const float* __restrict__ gam,
    const float* __restrict__ bet, const float* __restrict__ mu,
    const float* __restrict__ var)
{
    constexpr int OH   = H + 2 * PAD - KH + 1;
    constexpr int OW   = W + 2 * PAD - KW + 1;
    constexpr int K    = Cin * KH * KW;
    constexpr int KT   = (K + 31) / 32;
    constexpr int TPW  = MTB / 2;
    constexpr int NIMG = 1024;
    constexpr int NPIX = NIMG * OH * OW;

    __shared__ uint4 BshQ[256];                // 4 KB B chunk (fragment layout)
    __shared__ uint4 AshQ[MTB * 64];           // MTB KB A tiles (fragment layout)
    bf16* Bsh = (bf16*)BshQ;

    const int tid  = threadIdx.x;
    const int lane = tid & 31;
    const int wv   = tid >> 5;
    const int wn   = wv & 3;
    const int wm0  = (wv >> 2) * TPW;
    const int n0   = blockIdx.x * 64;
    const int mt0  = blockIdx.y * MTB;

    v8f acc[TPW];
#pragma unroll
    for (int t = 0; t < TPW; ++t) acc[t] = (v8f){};

    const int kr    = tid >> 3;                // 0..31: staged k row
    const int nbase = (tid & 7) * 8;           // 8 consecutive pixels
    const int grp16 = ((kr >> 3) & 1) * 16;
    const int krel  = (kr & 7) | ((kr >> 4) << 3);

    for (int kt = 0; kt < KT; ++kt) {
        __syncthreads();                       // LDS reads of prev iter done

#if HAS_ASYNC_LDS
        // ---- async DMA: wave wv (< MTB) copies its 1KB A m-tile into LDS ----
        if (wv < MTB) {
            const uint4* src = (const uint4*)Apack +
                ((size_t)(mt0 + wv) * KT + kt) * 64 + lane * 2;
            uint4* dst = AshQ + wv * 64 + lane * 2;
            async_copy16(src,     dst);
            async_copy16(src + 1, dst + 1);
        }
#endif
        // ---- stage 32x64 im2col chunk into LDS (B-fragment layout) ----
        {
            const int k    = kt * 32 + kr;
            const bool kok = (K % 32 == 0) ? true : (k < K);
            const int kw   = k % KW;
            const int kh   = (k / KW) % KH;
            const int cin  = k / (KW * KH);
#pragma unroll
            for (int u = 0; u < 8; ++u) {
                int nloc = nbase + u;
                int n = n0 + nloc;
                float val = 0.0f;
                if (kok && n < NPIX) {
                    int ow  = n % OW;
                    int tq  = n / OW;
                    int oh  = tq % OH;
                    int img = tq / OH;
                    int ih = oh + kh - PAD, iw = ow + kw - PAD;
                    bool ok = (PAD == 0) ? (ih < H && iw < W)
                                         : (ih >= 0 && ih < H && iw >= 0 && iw < W);
                    if (ok)
                        val = (float)in[(((size_t)img * Cin + cin) * H + ih) * W + iw];
                }
                int nsub = nloc >> 4, col = nloc & 15;
                Bsh[(nsub * 32 + grp16 + col) * 16 + krel] = (bf16)val;
            }
        }
#if HAS_ASYNC_LDS
        if (wv < MTB) wait_async0();           // DMA landed before the barrier
#endif
        __syncthreads();

        // ---- B fragment: two ds b128 loads (shared by this wave's m-tiles) --
        union { uint4 q[2]; v16bf v; } bu;
        const uint4* Bq = BshQ + wn * 64 + lane * 2;
        bu.q[0] = Bq[0];
        bu.q[1] = Bq[1];

#pragma unroll
        for (int t = 0; t < TPW; ++t) {
            union { uint4 q[2]; v16bf v; } au;
#if HAS_ASYNC_LDS
            const uint4* Aq = AshQ + (wm0 + t) * 64 + lane * 2;
#else
            const uint4* Aq = (const uint4*)Apack +
                ((size_t)(mt0 + wm0 + t) * KT + kt) * 64 + lane * 2;
#endif
            au.q[0] = Aq[0];
            au.q[1] = Aq[1];
            acc[t] = __builtin_amdgcn_wmma_f32_16x16x32_bf16(
                         false, au.v, false, bu.v, (short)0, acc[t], false, false);
        }
    }

    // ---- epilogue: bias + BN + ReLU, scatter to NCHW ----
    // C/D layout: lane 0-15: n=lane, rows 0..7; lane 16-31: n=lane-16, rows 8..15
    int nn = n0 + wn * 16 + (lane & 15);
    if (nn < NPIX) {
        int ow  = nn % OW;
        int tq  = nn / OW;
        int oh  = tq % OH;
        int img = tq / OH;
#pragma unroll
        for (int t = 0; t < TPW; ++t) {
            int mbase = (mt0 + wm0 + t) * 16 + ((lane >> 4) * 8);
#pragma unroll
            for (int r = 0; r < 8; ++r) {
                int c = mbase + r;
                if (Cout % 32 != 0 && c >= Cout) continue;
                float v = acc[t][r];
                if (FLAGS & 1) {
                    float inv = rsqrtf(var[c] + 1e-5f);
                    float s = gam[c] * inv;
                    v = v * s + (bias[c] - mu[c]) * s + bet[c];
                } else {
                    v = v + bias[c];
                }
                if (FLAGS & 2) v = fmaxf(v, 0.0f);
                size_t o = (((size_t)img * Cout + c) * OH + oh) * OW + ow;
                if (FLAGS & 4) outf[o] = v;
                else           outb[o] = (bf16)v;
            }
        }
    }
}

// ---------------------------------------------------------------------------
// MaxPool (2,1) along H on bf16 NCHW
// ---------------------------------------------------------------------------
__global__ void maxpool2(const bf16* __restrict__ in, bf16* __restrict__ out,
                         int N, int C, int OH, int W)
{
    int idx = blockIdx.x * blockDim.x + threadIdx.x;
    int total = N * C * OH * W;
    if (idx >= total) return;
    int w = idx % W;
    int t = idx / W;
    int oh = t % OH; t /= OH;
    int c = t % C;
    int n = t / C;
    size_t base = (((size_t)n * C + c) * (OH * 2) + oh * 2) * W + w;
    float a = (float)in[base];
    float b = (float)in[base + W];
    out[idx] = (bf16)fmaxf(a, b);
}

// ---------------------------------------------------------------------------
// Mean over HW: (1024,128,23,3) -> (1024,128)
// ---------------------------------------------------------------------------
__global__ void mean_hw(const bf16* __restrict__ in, bf16* __restrict__ out,
                        int NC, int HW)
{
    int idx = blockIdx.x * blockDim.x + threadIdx.x;
    if (idx >= NC) return;
    const bf16* p = in + (size_t)idx * HW;
    float s = 0.0f;
    for (int i = 0; i < HW; ++i) s += (float)p[i];
    out[idx] = (bf16)(s * (1.0f / HW));
}

// ---------------------------------------------------------------------------
extern "C" void kernel_launch(void* const* d_in, const int* in_sizes, int n_in,
                              void* d_out, int out_size, void* d_ws, size_t ws_size,
                              hipStream_t stream)
{
    (void)in_sizes; (void)n_in; (void)out_size; (void)ws_size;
    const float* x  = (const float*)d_in[0];
    auto P = [&](int layer, int which) -> const float* {
        return (const float*)d_in[1 + layer * 6 + which];
    };
    const float* wl = (const float*)d_in[43];
    const float* bl = (const float*)d_in[44];

    char* ws = (char*)d_ws;
    size_t off = 0;
    auto alloc = [&](size_t bytes) -> void* {
        void* p = ws + off;
        off += (bytes + 255) & ~(size_t)255;
        return p;
    };

    const int NIMG = 1024;
    bf16* win   = (bf16*)alloc((size_t)NIMG * 144 * 15 * sizeof(bf16));
    size_t actE = (size_t)NIMG * 32 * 144 * 15;
    bf16* bufA  = (bf16*)alloc(actE * sizeof(bf16));
    bf16* bufB  = (bf16*)alloc(actE * sizeof(bf16));
    bf16* hmean = (bf16*)alloc((size_t)NIMG * 128 * sizeof(bf16));

    // ---- weight prepack (7 convs + classifier) ----
    const int co[8] = {32,32,32,32,64,64,128,170};
    const int ci[8] = { 1,32,32,32,32,64, 64,128};
    const int kh[8] = { 3, 3, 3, 3, 3, 3, 12,  1};
    const int kw[8] = { 3, 3, 3, 3, 3, 3,  9,  1};
    unsigned* pk[8];
    for (int i = 0; i < 8; ++i) {
        int K  = ci[i] * kh[i] * kw[i];
        int KT = (K + 31) / 32;
        int MT = ((co[i] + 31) / 32) * 2;
        pk[i] = (unsigned*)alloc((size_t)MT * KT * 32 * 8 * sizeof(unsigned));
        const float* w = (i < 7) ? P(i, 0) : wl;
        int total = MT * KT * 32;
        pack_weights<<<(total + 255) / 256, 256, 0, stream>>>(w, pk[i], co[i], K, MT, KT);
    }

    // ---- windowed input for c1 ----
    {
        int total = NIMG * 144 * 15;
        build_window<<<(total + 255) / 256, 256, 0, stream>>>(x, win, total);
    }

    auto gx = [](int npix) { return (npix + 63) / 64; };

    // c1..c4: 144x15, pad 1, Cout 32 (MTB=2)
    conv_wmma<1,144,15,32,3,3,1,2,3><<<dim3(gx(NIMG*144*15),1),256,0,stream>>>(
        win,  pk[0], bufA, nullptr, P(0,1),P(0,2),P(0,3),P(0,4),P(0,5));
    conv_wmma<32,144,15,32,3,3,1,2,3><<<dim3(gx(NIMG*144*15),1),256,0,stream>>>(
        bufA, pk[1], bufB, nullptr, P(1,1),P(1,2),P(1,3),P(1,4),P(1,5));
    conv_wmma<32,144,15,32,3,3,1,2,3><<<dim3(gx(NIMG*144*15),1),256,0,stream>>>(
        bufB, pk[2], bufA, nullptr, P(2,1),P(2,2),P(2,3),P(2,4),P(2,5));
    conv_wmma<32,144,15,32,3,3,1,2,3><<<dim3(gx(NIMG*144*15),1),256,0,stream>>>(
        bufA, pk[3], bufB, nullptr, P(3,1),P(3,2),P(3,3),P(3,4),P(3,5));

    // maxpool: (1024,32,144,15) -> (1024,32,72,15)
    {
        int total = NIMG * 32 * 72 * 15;
        maxpool2<<<(total + 255) / 256, 256, 0, stream>>>(bufB, bufA, NIMG, 32, 72, 15);
    }

    // c5: 72x15 -> 70x13 (Cout 64, MTB=4) ; c6: 70x13 -> 68x11
    conv_wmma<32,72,15,64,3,3,0,4,3><<<dim3(gx(NIMG*70*13),1),256,0,stream>>>(
        bufA, pk[4], bufB, nullptr, P(4,1),P(4,2),P(4,3),P(4,4),P(4,5));
    conv_wmma<64,70,13,64,3,3,0,4,3><<<dim3(gx(NIMG*68*11),1),256,0,stream>>>(
        bufB, pk[5], bufA, nullptr, P(5,1),P(5,2),P(5,3),P(5,4),P(5,5));

    // maxpool: (1024,64,68,11) -> (1024,64,34,11)
    {
        int total = NIMG * 64 * 34 * 11;
        maxpool2<<<(total + 255) / 256, 256, 0, stream>>>(bufA, bufB, NIMG, 64, 34, 11);
    }

    // c7: 34x11, 12x9 -> 23x3 (Cout 128, MTB=8, K=6912 -> the big GEMM)
    conv_wmma<64,34,11,128,12,9,0,8,3><<<dim3(gx(NIMG*23*3),1),256,0,stream>>>(
        bufB, pk[6], bufA, nullptr, P(6,1),P(6,2),P(6,3),P(6,4),P(6,5));

    // mean over 23x3 (commutes with the 1x1 conv): (1024,128,69) -> (1024,128)
    {
        int NC = NIMG * 128;
        mean_hw<<<(NC + 255) / 256, 256, 0, stream>>>(bufA, hmean, NC, 23 * 3);
    }

    // classifier: 170(->192)x1024x128 GEMM, bias only, f32 out -> d_out
    conv_wmma<128,1,1,170,1,1,0,4,4><<<dim3(gx(NIMG),3),256,0,stream>>>(
        hmean, pk[7], nullptr, (float*)d_out, bl, nullptr, nullptr, nullptr, nullptr);
}
